// CausalSelfAttention_71571335020942
// MI455X (gfx1250) — matrix operand order repaired and checked by reading, not compiled, not run on hardware
//
#include <hip/hip_runtime.h>

typedef __attribute__((ext_vector_type(16))) __bf16 v16bf;
typedef __attribute__((ext_vector_type(8)))  __bf16 v8bf;
typedef __attribute__((ext_vector_type(8)))  float  v8f;

constexpr int B_ = 4, S_ = 2048, C_ = 2048, H_ = 16, D_ = 128;
constexpr int NTOK = B_ * S_;     // 8192 tokens
constexpr int NQKV = 3 * C_;      // 6144

#define DEV __device__ __forceinline__

DEV v8f wmma_bf16(v16bf a, v16bf b, v8f c) {
  // D = A(16x32) * B(32x16) + C, f32 accumulate
  return __builtin_amdgcn_wmma_f32_16x16x32_bf16(
      /*neg_a=*/false, a, /*neg_b=*/false, b,
      /*c_mod=*/(short)0, c, /*reuse_a=*/false, /*reuse_b=*/false);
}

// A-fragment (16x32 bf16): lane holds row m=lane&15.
// a[i] <-> k = 16*(lane>=16) + i  => two contiguous 16B chunks.
DEV v16bf afrag(const __bf16* rowp, int hi8) {
  v8bf lo = *reinterpret_cast<const v8bf*>(rowp + 2 * hi8);
  v8bf hh = *reinterpret_cast<const v8bf*>(rowp + 2 * hi8 + 8);
  return __builtin_shufflevector(lo, hh, 0,1,2,3,4,5,6,7,8,9,10,11,12,13,14,15);
}

// B-fragment (32x16 bf16): lane holds col n=lane&15; b[i] <-> k = 16*(lane>=16)+i
// => one contiguous 32B load when the K axis is contiguous in memory.
DEV v16bf bfrag(const __bf16* p) { return *reinterpret_cast<const v16bf*>(p); }

// Butterfly reductions across the 16 lanes of a half-wave (C-layout row lives there).
DEV float redmax16(float v) {
  v = fmaxf(v, __shfl_xor(v, 1, 32));
  v = fmaxf(v, __shfl_xor(v, 2, 32));
  v = fmaxf(v, __shfl_xor(v, 4, 32));
  v = fmaxf(v, __shfl_xor(v, 8, 32));
  return v;
}
DEV float redsum16(float v) {
  v += __shfl_xor(v, 1, 32);
  v += __shfl_xor(v, 2, 32);
  v += __shfl_xor(v, 4, 32);
  v += __shfl_xor(v, 8, 32);
  return v;
}

// ---------------- conversion kernels ----------------
__global__ void cvt_f32_bf16(const float* __restrict__ src, __bf16* __restrict__ dst, int n) {
  int i = blockIdx.x * blockDim.x + threadIdx.x;
  if (i < n) dst[i] = (__bf16)src[i];
}

// src is [K][N] f32 row-major; dst is [N][K] bf16 (so WMMA B-frags read contiguous K)
__global__ void cvtT_f32_bf16(const float* __restrict__ src, __bf16* __restrict__ dst,
                              int K, int N) {
  int i = blockIdx.x * blockDim.x + threadIdx.x;
  int total = K * N;
  if (i < total) {
    int n = i / K, k = i - n * K;
    dst[i] = (__bf16)src[(size_t)k * N + n];
  }
}

// ---------------- QKV GEMM: [8192 x 2048] @ [2048 x 6144] ----------------
// One wave per 64x64 output tile (4x4 WMMA tiles, register-blocked):
// 16 WMMAs per 8 fragment loads. Scatter into Q/K [B,H,S,D] and V^T [B,H,D,S].
__global__ void __launch_bounds__(256) gemm_qkv(
    const __bf16* __restrict__ A, const __bf16* __restrict__ BT,
    const float* __restrict__ bias,
    __bf16* __restrict__ Qo, __bf16* __restrict__ Ko, __bf16* __restrict__ Vt) {
  constexpr int K = C_;
  constexpr int TN = NQKV / 64;   // 96 tiles along N
  int wid  = blockIdx.x * (blockDim.x >> 5) + (threadIdx.x >> 5);
  int lane = threadIdx.x & 31;
  int tm = wid / TN, tn = wid - tm * TN;
  int hi = lane >> 4, nn = lane & 15;
  int hi8 = hi * 8, hi16 = hi * 16;

  const __bf16* aps[4];
  const __bf16* bps[4];
#pragma unroll
  for (int mi = 0; mi < 4; ++mi)
    aps[mi] = A + (size_t)(tm * 64 + mi * 16 + nn) * K;
#pragma unroll
  for (int ni = 0; ni < 4; ++ni)
    bps[ni] = BT + (size_t)(tn * 64 + ni * 16 + nn) * K + hi16;

  v8f zero = {};
  v8f acc[4][4];
#pragma unroll
  for (int mi = 0; mi < 4; ++mi)
#pragma unroll
    for (int ni = 0; ni < 4; ++ni) acc[mi][ni] = zero;

  for (int kb = 0; kb < K; kb += 32) {
    v16bf af[4], bfv[4];
#pragma unroll
    for (int mi = 0; mi < 4; ++mi) af[mi] = afrag(aps[mi] + kb, hi8);
#pragma unroll
    for (int ni = 0; ni < 4; ++ni) bfv[ni] = bfrag(bps[ni] + kb);
#pragma unroll
    for (int mi = 0; mi < 4; ++mi)
#pragma unroll
      for (int ni = 0; ni < 4; ++ni)
        acc[mi][ni] = wmma_bf16(af[mi], bfv[ni], acc[mi][ni]);
  }

#pragma unroll
  for (int ni = 0; ni < 4; ++ni) {
    int col = tn * 64 + ni * 16 + nn;
    float bs = bias[col];
#pragma unroll
    for (int mi = 0; mi < 4; ++mi)
#pragma unroll
      for (int r = 0; r < 8; ++r) {
        int tok = tm * 64 + mi * 16 + r + hi8;     // token index
        int bb = tok >> 11, ss = tok & (S_ - 1);
        __bf16 hv = (__bf16)(acc[mi][ni][r] + bs);
        if (col < C_) {
          int h = col >> 7, d = col & 127;
          Qo[(((size_t)bb * H_ + h) * S_ + ss) * D_ + d] = hv;
        } else if (col < 2 * C_) {
          int cc = col - C_; int h = cc >> 7, d = cc & 127;
          Ko[(((size_t)bb * H_ + h) * S_ + ss) * D_ + d] = hv;
        } else {
          int cc = col - 2 * C_; int h = cc >> 7, d = cc & 127;
          Vt[(((size_t)bb * H_ + h) * D_ + d) * S_ + ss] = hv;   // transposed V
        }
      }
  }
}

// ---------------- flash attention: one wave per (b,h, 16-query tile) ----------------
// Softmax row statistics fully in registers via half-wave shuffle butterflies;
// LDS used only for the C-layout -> A-layout transpose of P (bf16, 1KB/wave).
__global__ void __launch_bounds__(256) attn_fwd(
    const __bf16* __restrict__ Q, const __bf16* __restrict__ Kd,
    const __bf16* __restrict__ Vt, __bf16* __restrict__ Y) {
  __shared__ alignas(32) __bf16 pb[8][16][32];   // P row-major per wave

  int w    = threadIdx.x >> 5;
  int lane = threadIdx.x & 31;
  int wid  = blockIdx.x * 8 + w;
  int bh = wid >> 7;          // / (S/16) q-tiles
  int qt = wid & 127;
  int hi = lane >> 4, nn = lane & 15;
  int hi8 = hi * 8, hi16 = hi * 16;

  const __bf16* Qb = Q  + (size_t)bh * S_ * D_;
  const __bf16* Kb = Kd + (size_t)bh * S_ * D_;
  const __bf16* Vb = Vt + (size_t)bh * D_ * S_;   // [D][S]

  // preload Q fragments for the 4 K-steps covering D=128
  v16bf qf[4];
  {
    const __bf16* qp = Qb + (size_t)(qt * 16 + nn) * D_;
#pragma unroll
    for (int t = 0; t < 4; ++t) qf[t] = afrag(qp + t * 32, hi8);
  }

  v8f zero = {};
  v8f o[8];
#pragma unroll
  for (int t = 0; t < 8; ++t) o[t] = zero;

  // per-row online-softmax state (rows r+8*hi), replicated across the half-wave
  float mst[8], lst[8];
#pragma unroll
  for (int r = 0; r < 8; ++r) { mst[r] = -1e30f; lst[r] = 0.0f; }

  const float scale = 0.08838834764831845f;   // 1/sqrt(128)
  int kend = qt * 16 + 16;                    // causal upper bound (exclusive)

  for (int kb = 0; kb < kend; kb += 32) {
    // ---- S = Q K^T for 32 keys (two 16-col tiles) ----
    v8f c0 = zero, c1 = zero;
#pragma unroll
    for (int t = 0; t < 4; ++t) {
      v16bf k0 = bfrag(Kb + (size_t)(kb + nn) * D_ + t * 32 + hi16);
      c0 = wmma_bf16(qf[t], k0, c0);
      v16bf k1 = bfrag(Kb + (size_t)(kb + 16 + nn) * D_ + t * 32 + hi16);
      c1 = wmma_bf16(qf[t], k1, c1);
    }
    // ---- scale, causal mask, online softmax in registers ----
    float alq[8];
#pragma unroll
    for (int r = 0; r < 8; ++r) {
      int row = r + hi8;                    // query row within tile
      int qg = qt * 16 + row;               // global query index
      float s0 = c0[r] * scale;
      float s1 = c1[r] * scale;
      if (kb + nn > qg)      s0 = -1e30f;
      if (kb + 16 + nn > qg) s1 = -1e30f;
      float mt   = redmax16(fmaxf(s0, s1));
      float mnew = fmaxf(mst[r], mt);
      float al   = __expf(mst[r] - mnew);
      float p0   = __expf(s0 - mnew);
      float p1   = __expf(s1 - mnew);
      float rs   = redsum16(p0 + p1);
      lst[r] = lst[r] * al + rs;
      mst[r] = mnew;
      alq[r] = al;
      pb[w][row][nn]      = (__bf16)p0;
      pb[w][row][16 + nn] = (__bf16)p1;
    }
    // ---- rescale accumulator ----
#pragma unroll
    for (int t = 0; t < 8; ++t)
#pragma unroll
      for (int r = 0; r < 8; ++r) o[t][r] *= alq[r];
    // ---- O += P V : P reloaded from LDS in A-layout, V^T gives contiguous B-frags ----
    asm volatile("s_wait_dscnt 0" ::: "memory");
    v16bf pf = afrag(&pb[w][nn][0], hi8);
#pragma unroll
    for (int t = 0; t < 8; ++t) {
      v16bf vf = bfrag(Vb + (size_t)(t * 16 + nn) * S_ + kb + hi16);
      o[t] = wmma_bf16(pf, vf, o[t]);
    }
  }

  float li[8];
#pragma unroll
  for (int r = 0; r < 8; ++r) li[r] = 1.0f / lst[r];

  int b = bh >> 4, h = bh & 15;
#pragma unroll
  for (int t = 0; t < 8; ++t)
#pragma unroll
    for (int r = 0; r < 8; ++r) {
      int row = r + hi8;
      size_t tok = (size_t)b * S_ + qt * 16 + row;
      Y[tok * C_ + h * D_ + t * 16 + nn] = (__bf16)(o[t][r] * li[r]);
    }
}

// ---------------- projection GEMM: [8192 x 2048] @ [2048 x 2048] -> f32 ----------------
// Same 64x64-per-wave register blocking; f32 output with bias.
__global__ void __launch_bounds__(256) gemm_proj(
    const __bf16* __restrict__ A, const __bf16* __restrict__ BT,
    const float* __restrict__ bias, float* __restrict__ O) {
  constexpr int K = C_;
  constexpr int TN = C_ / 64;     // 32
  int wid  = blockIdx.x * (blockDim.x >> 5) + (threadIdx.x >> 5);
  int lane = threadIdx.x & 31;
  int tm = wid / TN, tn = wid - tm * TN;
  int hi = lane >> 4, nn = lane & 15;
  int hi8 = hi * 8, hi16 = hi * 16;

  const __bf16* aps[4];
  const __bf16* bps[4];
#pragma unroll
  for (int mi = 0; mi < 4; ++mi)
    aps[mi] = A + (size_t)(tm * 64 + mi * 16 + nn) * K;
#pragma unroll
  for (int ni = 0; ni < 4; ++ni)
    bps[ni] = BT + (size_t)(tn * 64 + ni * 16 + nn) * K + hi16;

  v8f zero = {};
  v8f acc[4][4];
#pragma unroll
  for (int mi = 0; mi < 4; ++mi)
#pragma unroll
    for (int ni = 0; ni < 4; ++ni) acc[mi][ni] = zero;

  for (int kb = 0; kb < K; kb += 32) {
    v16bf af[4], bfv[4];
#pragma unroll
    for (int mi = 0; mi < 4; ++mi) af[mi] = afrag(aps[mi] + kb, hi8);
#pragma unroll
    for (int ni = 0; ni < 4; ++ni) bfv[ni] = bfrag(bps[ni] + kb);
#pragma unroll
    for (int mi = 0; mi < 4; ++mi)
#pragma unroll
      for (int ni = 0; ni < 4; ++ni)
        acc[mi][ni] = wmma_bf16(af[mi], bfv[ni], acc[mi][ni]);
  }

#pragma unroll
  for (int ni = 0; ni < 4; ++ni) {
    int col = tn * 64 + ni * 16 + nn;
    float bs = bias[col];
#pragma unroll
    for (int mi = 0; mi < 4; ++mi)
#pragma unroll
      for (int r = 0; r < 8; ++r) {
        int tok = tm * 64 + mi * 16 + r + hi8;
        O[(size_t)tok * C_ + col] = acc[mi][ni][r] + bs;
      }
  }
}

// ---------------- launch ----------------
extern "C" void kernel_launch(void* const* d_in, const int* in_sizes, int n_in,
                              void* d_out, int out_size, void* d_ws, size_t ws_size,
                              hipStream_t stream) {
  (void)in_sizes; (void)n_in; (void)out_size; (void)ws_size;
  const float* x      = (const float*)d_in[0];
  const float* w_qkv  = (const float*)d_in[1];
  const float* b_qkv  = (const float*)d_in[2];
  const float* w_proj = (const float*)d_in[3];
  const float* b_proj = (const float*)d_in[4];
  float* out = (float*)d_out;

  char* ws = (char*)d_ws;
  size_t off = 0;
  __bf16* xb     = (__bf16*)(ws + off); off += (size_t)NTOK * C_ * 2;   // 33.5 MB
  __bf16* wqkvT  = (__bf16*)(ws + off); off += (size_t)NQKV * C_ * 2;   // 25 MB
  __bf16* wprojT = (__bf16*)(ws + off); off += (size_t)C_ * C_ * 2;     // 8.4 MB
  __bf16* Qb     = (__bf16*)(ws + off); off += (size_t)NTOK * C_ * 2;   // 33.5 MB
  __bf16* Kb     = (__bf16*)(ws + off); off += (size_t)NTOK * C_ * 2;
  __bf16* Vt     = (__bf16*)(ws + off); off += (size_t)NTOK * C_ * 2;
  __bf16* Yb     = (__bf16*)(ws + off); off += (size_t)NTOK * C_ * 2;

  // 1) precision conversion + weight transposes
  cvt_f32_bf16 <<<(NTOK * C_) / 256, 256, 0, stream>>>(x, xb, NTOK * C_);
  cvtT_f32_bf16<<<(C_ * NQKV) / 256, 256, 0, stream>>>(w_qkv, wqkvT, C_, NQKV);
  cvtT_f32_bf16<<<(C_ * C_)   / 256, 256, 0, stream>>>(w_proj, wprojT, C_, C_);

  // 2) QKV projection: 128 x 96 waves of 64x64 tiles, 8 waves/block
  gemm_qkv<<<(128 * 96) / 8, 256, 0, stream>>>(xb, wqkvT, b_qkv, Qb, Kb, Vt);

  // 3) flash attention (8192 waves: B*H * S/16)
  attn_fwd<<<(B_ * H_ * (S_ / 16)) / 8, 256, 0, stream>>>(Qb, Kb, Vt, Yb);

  // 4) output projection: 128 x 32 waves of 64x64 tiles, f32 out
  gemm_proj<<<(128 * 32) / 8, 256, 0, stream>>>(Yb, wprojT, b_proj, out);
}
